// MLP_352187319085
// MI455X (gfx1250) — compile-verified, matrix operand order checked by reading, not run on hardware
//
#include <hip/hip_runtime.h>

#define D 512           // D_IN == D_OUT == 512
#define APITCH 520      // LDS row pitch in bf16 elems: 1040 B (16B-aligned, bank-spread)
#define MTILE 32        // rows per block (100000 % 32 == 0)

typedef __attribute__((ext_vector_type(16))) __bf16 v16bf;
typedef __attribute__((ext_vector_type(8)))  __bf16 v8bf;
typedef __attribute__((ext_vector_type(4)))  __bf16 v4bf;
typedef __attribute__((ext_vector_type(8)))  float  v8f;

// ---------------------------------------------------------------------------
// Kernel 0: out[i, d] = bias[d]   (out is poisoned by the harness)
// ---------------------------------------------------------------------------
__global__ __launch_bounds__(256)
void gcn_init_bias(float* __restrict__ out, const float* __restrict__ bias,
                   size_t total) {
  size_t i = (size_t)blockIdx.x * blockDim.x + threadIdx.x;
  size_t stride = (size_t)gridDim.x * blockDim.x;
  for (; i < total; i += stride)
    out[i] = bias[i & (D - 1)];
}

// ---------------------------------------------------------------------------
// Kernel 1: one-time swizzle of W[512,512] f32 into bf16 B-fragment layout.
// Record (n_tile, k_step, lane) = the 16 bf16 values lane `lane` feeds into
// v_wmma_f32_16x16x32_bf16 for B-tile (k_step*32 .. +31) x (n_tile*16 .. +15):
//   n = n_tile*16 + (lane&15);  K-pair base kb = (v<4 ? 2v : 2v+8) + 8*(lane>>4)
// Stored contiguously: 32 B per lane -> GEMM reads it as 2 coalesced b128s.
// ---------------------------------------------------------------------------
__global__ __launch_bounds__(32)
void gcn_swizzle_w(const float* __restrict__ W, __bf16* __restrict__ Wsw) {
  const int n_tile = blockIdx.x;       // 0..31
  const int ks     = blockIdx.y;       // 0..15
  const int lane   = threadIdx.x;      // 0..31
  const int half   = lane >> 4;
  const int ncol   = n_tile * 16 + (lane & 15);
  const int k0     = ks * 32;

  v16bf b;
#pragma unroll
  for (int v = 0; v < 8; ++v) {
    const int kb = ((v < 4) ? (2 * v) : (2 * v + 8)) + half * 8;
    b[2 * v]     = (__bf16)W[(size_t)(k0 + kb) * D + ncol];
    b[2 * v + 1] = (__bf16)W[(size_t)(k0 + kb + 1) * D + ncol];
  }
  *(v16bf*)(Wsw + ((((size_t)n_tile * 16) + ks) * 32 + lane) * 16) = b;
}

// ---------------------------------------------------------------------------
// Kernel 2: support[M, 512] = X[M, 512] @ W[512, 512]  (bf16 WMMA, f32 accum)
// Block = 256 threads = 8 waves; block tile = 32 rows x 128 cols.
// Phase 1: cooperatively stage the 32x512 X tile into LDS as bf16 (33 KB).
// Phase 2: per wave, one n_tile, TWO 16-row M-subtiles sharing each B
//          fragment: per K-step = 2x global_load_b128 (B) + 4x ds_load_b128
//          (A, immediate offsets) + 2 WMMA  ->  1 VMEM + 2 DS per WMMA.
// Requires M % 32 == 0 (true: 100000) and N = 512.
// ---------------------------------------------------------------------------
__global__ __launch_bounds__(256)
void gcn_gemm_wmma(const float* __restrict__ X, const __bf16* __restrict__ Wsw,
                   float* __restrict__ support, int M) {
  __shared__ __bf16 sX[MTILE * APITCH];   // 33.3 KB

  const int tid  = threadIdx.x;
  const int lane = tid & 31;
  const int wave = tid >> 5;
  const int half = lane >> 4;          // 0 = lanes 0-15, 1 = lanes 16-31
  const int l16  = lane & 15;

  const int m0 = blockIdx.y * MTILE;

  // ---- Phase 1: X tile -> LDS (bf16). 4096 float4 chunks, 16 per thread ----
#pragma unroll
  for (int i = 0; i < 16; ++i) {
    const int idx = tid + i * 256;           // 0..4095
    const int row = idx >> 7;                // 0..31
    const int c4  = (idx & 127) * 4;         // 0..508
    const float4 f = *(const float4*)(X + (size_t)(m0 + row) * D + c4);
    v4bf c;
    c[0] = (__bf16)f.x; c[1] = (__bf16)f.y;
    c[2] = (__bf16)f.z; c[3] = (__bf16)f.w;
    *(v4bf*)(sX + row * APITCH + c4) = c;
  }
  __syncthreads();

  // ---- Phase 2: 16 K-steps, 2 M-subtiles per wave ----
  const int n_tile = blockIdx.x * 8 + wave;            // 0..31
  // per-n_tile record: 16 ks * 32 lanes * 16 bf16 = 8192 elems
  const __bf16* __restrict__ wbase =
      Wsw + (size_t)n_tile * 8192 + (size_t)lane * 16;

  const __bf16* __restrict__ arow = sX + l16 * APITCH + half * 8;

  v8f acc0 = {};
  v8f acc1 = {};
#pragma unroll 2
  for (int ks = 0; ks < 16; ++ks) {
    const int k0 = ks * 32;
    // B fragment: contiguous 32 B per lane (coalesced 2x b128), reused twice
    const v16bf b = *(const v16bf*)(wbase + (size_t)ks * 512);

    // A fragment, subtile 0 (rows m0 .. m0+15)
    const v8bf alo0 = *(const v8bf*)(arow + k0);
    const v8bf ahi0 = *(const v8bf*)(arow + k0 + 16);
    const v16bf a0 = __builtin_shufflevector(alo0, ahi0,
        0, 1, 2, 3, 4, 5, 6, 7, 8, 9, 10, 11, 12, 13, 14, 15);
    acc0 = __builtin_amdgcn_wmma_f32_16x16x32_bf16(
        false, a0, false, b, (short)0, acc0, false, false);

    // A fragment, subtile 1 (rows m0+16 .. m0+31)
    const v8bf alo1 = *(const v8bf*)(arow + 16 * APITCH + k0);
    const v8bf ahi1 = *(const v8bf*)(arow + 16 * APITCH + k0 + 16);
    const v16bf a1 = __builtin_shufflevector(alo1, ahi1,
        0, 1, 2, 3, 4, 5, 6, 7, 8, 9, 10, 11, 12, 13, 14, 15);
    acc1 = __builtin_amdgcn_wmma_f32_16x16x32_bf16(
        false, a1, false, b, (short)0, acc1, false, false);
  }

  // ---- D stores: VGPR r -> row (subtile base + 8*half + r), col n_tile*16+l16
  float* __restrict__ srow =
      support + (size_t)(m0 + half * 8) * D + n_tile * 16 + l16;
#pragma unroll
  for (int r = 0; r < 8; ++r)
    srow[(size_t)r * D] = acc0[r];
  float* __restrict__ srow1 = srow + (size_t)16 * D;
#pragma unroll
  for (int r = 0; r < 8; ++r)
    srow1[(size_t)r * D] = acc1[r];
}

// ---------------------------------------------------------------------------
// Kernel 3: COO scatter: out[row[e], :] += val[e] * support[col[e], :]
// 128 lanes per edge, float4 gather, 4 f32 global atomics (no-return, L2).
// support (205 MB) is ~L2-resident (192 MB) with avg degree 32 reuse.
// ---------------------------------------------------------------------------
__global__ __launch_bounds__(256)
void gcn_edge_scatter(const int* __restrict__ erow, const int* __restrict__ ecol,
                      const float* __restrict__ eval,
                      const float* __restrict__ support,
                      float* __restrict__ out, int E) {
  const int slot = threadIdx.x >> 7;          // 2 edges per 256-thread block
  const int d    = (threadIdx.x & 127) * 4;   // 128 lanes * 4 floats = 512 cols
  const long long estride = (long long)gridDim.x * 2;

  for (long long e = (long long)blockIdx.x * 2 + slot; e < E; e += estride) {
    const int   src = ecol[e];
    const int   dst = erow[e];
    const float v   = eval[e];

    const float4 s = *(const float4*)(support + (size_t)src * D + d);
    float* o = out + (size_t)dst * D + d;
    atomicAdd(o + 0, v * s.x);
    atomicAdd(o + 1, v * s.y);
    atomicAdd(o + 2, v * s.z);
    atomicAdd(o + 3, v * s.w);
  }
}

// ---------------------------------------------------------------------------
// Launch: inputs = {input, edge_row, edge_col, edge_val, weight, bias}
// d_ws layout: [0, M*512*4)          : support (f32)
//              [M*512*4, +512*512*2) : swizzled bf16 W
// ---------------------------------------------------------------------------
extern "C" void kernel_launch(void* const* d_in, const int* in_sizes, int n_in,
                              void* d_out, int out_size, void* d_ws, size_t ws_size,
                              hipStream_t stream) {
  const float* X    = (const float*)d_in[0];
  const int*   erow = (const int*)  d_in[1];
  const int*   ecol = (const int*)  d_in[2];
  const float* ev   = (const float*)d_in[3];
  const float* W    = (const float*)d_in[4];
  const float* bias = (const float*)d_in[5];
  float* out = (float*)d_out;

  const int M = in_sizes[0] / D;   // 100000 nodes (multiple of 32)
  const int E = in_sizes[1];       // 3200000 edges

  float*  support = (float*)d_ws;
  __bf16* Wsw = (__bf16*)((char*)d_ws + (size_t)M * D * sizeof(float));

  // 1) swizzle W -> bf16 B-fragment layout (one-time, tiny)
  gcn_swizzle_w<<<dim3(32, 16), 32, 0, stream>>>(W, Wsw);

  // 2) out = bias
  gcn_init_bias<<<2048, 256, 0, stream>>>(out, bias, (size_t)M * D);

  // 3) support = X @ W  (WMMA bf16)
  gcn_gemm_wmma<<<dim3(4, M / MTILE), 256, 0, stream>>>(X, Wsw, support, M);

  // 4) scatter-add over edges
  long long want = ((long long)E + 1) / 2;
  int nblk = (want > (1 << 20)) ? (1 << 20) : (int)want;
  gcn_edge_scatter<<<nblk, 256, 0, stream>>>(erow, ecol, ev, support, out, E);
}
